// HierarchicalAttention_39376260169969
// MI455X (gfx1250) — compile-verified, matrix operand order checked by reading, not run on hardware
//
#include <hip/hip_runtime.h>

// ---------------------------------------------------------------------------
// Types
// ---------------------------------------------------------------------------
typedef _Float16 half_t;
typedef _Float16 v4h  __attribute__((ext_vector_type(4)));
typedef _Float16 v8h  __attribute__((ext_vector_type(8)));
typedef _Float16 v16h __attribute__((ext_vector_type(16)));
typedef float    v8f  __attribute__((ext_vector_type(8)));
typedef int      v4i_g __attribute__((vector_size(16)));   // matches async-LDS builtin

#define BS    8
#define QLEN  128
#define S1    32
#define S2    64
#define HDIM  768
#define NH    12
#define DH    64
#define SCALE 0.125f  // 1/sqrt(64)

#define __AS3 __attribute__((address_space(3)))
#define __AS1 __attribute__((address_space(1)))

#if __has_builtin(__builtin_amdgcn_global_load_async_to_lds_b128)
#define HAVE_ASYNC_LDS 1
#else
#define HAVE_ASYNC_LDS 0
#endif

// 16-byte global -> LDS copy; async (ASYNCcnt-tracked, VGPR-bypassing) when
// the gfx1250 builtin is available, else synchronous fallback.
__device__ __forceinline__ void cp16_g2l(const half_t* g, half_t* l) {
#if HAVE_ASYNC_LDS
    __builtin_amdgcn_global_load_async_to_lds_b128(
        (__AS1 v4i_g*)const_cast<half_t*>(g), (__AS3 v4i_g*)l, 0, 0);
#else
    *reinterpret_cast<v8h*>(l) = *reinterpret_cast<const v8h*>(g);
#endif
}

template <int N>
__device__ __forceinline__ void wait_async() {
#if HAVE_ASYNC_LDS
#if __has_builtin(__builtin_amdgcn_s_wait_asynccnt)
    __builtin_amdgcn_s_wait_asynccnt(N);
#else
    asm volatile("s_wait_asynccnt %0" ::"i"(N) : "memory");
#endif
#endif
}

// ---------------------------------------------------------------------------
// Fragment helpers (wave32, 16x16x32 f16 WMMA)
// A frag: lane&15 = M row; halfsel = lane>>4 selects K {0..7}/{8..15}, +16 for
//         the second 8 halves.  B frag identical with lane&15 = N column and
//         the source stored [N][K] (i.e. B transposed / column-major).
// ---------------------------------------------------------------------------
__device__ __forceinline__ v16h ld_frag(const half_t* lo) {
    union { v16h v; v8h h[2]; } u;
    u.h[0] = *reinterpret_cast<const v8h*>(lo);
    u.h[1] = *reinterpret_cast<const v8h*>(lo + 16);
    return u.v;
}

__device__ __forceinline__ v8f wmma_f16(v16h a, v16h b, v8f c) {
    return __builtin_amdgcn_wmma_f32_16x16x32_f16(false, a, false, b,
                                                  (short)0, c, false, false);
}

__device__ __forceinline__ v8f zero8() {
    v8f z = {0.f, 0.f, 0.f, 0.f, 0.f, 0.f, 0.f, 0.f};
    return z;
}

// ---------------------------------------------------------------------------
// Kernel 0: convert fp32 weight [K][N] -> f16 transposed [N][K]
// ---------------------------------------------------------------------------
__global__ __launch_bounds__(256) void cvt_wt(const float* __restrict__ W,
                                              half_t* __restrict__ Wt) {
    int i = blockIdx.x * 256 + threadIdx.x;   // 0 .. 768*768-1
    int k = i / HDIM;
    int n = i % HDIM;                          // coalesced read along n
    Wt[(size_t)n * HDIM + k] = (half_t)W[(size_t)k * HDIM + n];
}

// ---------------------------------------------------------------------------
// Kernel 1: GEMM  C[M][768](f16) = A[M][768] @ W[768][768] + bias
//   Wt is f16 pre-transposed [N][K].  Block tile 128x64, 8 waves, wave tile
//   32x32 (2x2 WMMA accumulators), K-step 64, double-buffered async staging.
// ---------------------------------------------------------------------------
template <bool A_IS_FLOAT>
__global__ __launch_bounds__(256) void gemm768(const void* __restrict__ Ain,
                                               const half_t* __restrict__ Wt,
                                               const float* __restrict__ bias,
                                               half_t* __restrict__ C) {
    __shared__ half_t As[2][128 * 64];   // 2 x 16 KB
    __shared__ half_t Bs[2][64 * 64];    // 2 x  8 KB

    const int tid  = threadIdx.x;
    const int lane = tid & 31;
    const int w    = tid >> 5;        // wave 0..7
    const int hs   = lane >> 4;       // halfsel
    const int lr   = lane & 15;
    const int wm   = w >> 1;          // 0..3 -> M sub-tile of 32
    const int wn   = w & 1;           // 0..1 -> N sub-tile of 32
    const int m0   = blockIdx.x * 128;
    const int n0   = blockIdx.y * 64;

    v8f acc[2][2];
    acc[0][0] = zero8(); acc[0][1] = zero8();
    acc[1][0] = zero8(); acc[1][1] = zero8();

    // async instructions issued per thread per stage (pipeline depth)
    constexpr int NW = A_IS_FLOAT ? 2 : 6;

    auto stage = [&](int kt, int buf) {
        const int k0 = kt * 64;
        half_t* As_ = &As[buf][0];
        half_t* Bs_ = &Bs[buf][0];
        if constexpr (A_IS_FLOAT) {
            // fp32 -> f16 conversion path (VALU), 32 elements / thread
            const float* A = (const float*)Ain;
            #pragma unroll
            for (int j = 0; j < 8; ++j) {
                const int c   = tid + 256 * j;   // 0..2047 float4 chunks
                const int row = c >> 4;          // 16 chunks per 64-float row
                const int col = (c & 15) * 4;
                float4 f = *reinterpret_cast<const float4*>(
                    A + (size_t)(m0 + row) * HDIM + k0 + col);
                v4h hv = { (half_t)f.x, (half_t)f.y, (half_t)f.z, (half_t)f.w };
                *reinterpret_cast<v4h*>(As_ + row * 64 + col) = hv;
            }
        } else {
            // f16 A: async global->LDS, 4 x b128 / thread
            const half_t* A = (const half_t*)Ain;
            #pragma unroll
            for (int j = 0; j < 4; ++j) {
                const int c   = tid + 256 * j;   // 0..1023 16B chunks
                const int row = c >> 3;          // 8 chunks per 64-half row
                const int col = (c & 7) * 8;
                cp16_g2l(A + (size_t)(m0 + row) * HDIM + k0 + col,
                         As_ + row * 64 + col);
            }
        }
        // B tile 64x64 ([N][K]) straight async copy from Wt, 2 x b128 / thread
        #pragma unroll
        for (int j = 0; j < 2; ++j) {
            const int c   = tid + 256 * j;       // 0..511 16B chunks
            const int row = c >> 3;
            const int col = (c & 7) * 8;
            cp16_g2l(Wt + (size_t)(n0 + row) * HDIM + k0 + col,
                     Bs_ + row * 64 + col);
        }
    };

    stage(0, 0);
    for (int kt = 0; kt < 12; ++kt) {
        const int cur = kt & 1;
        if (kt < 11) {
            stage(kt + 1, cur ^ 1);
            wait_async<NW>();    // current tile done; next batch in flight
        } else {
            wait_async<0>();
        }
        __syncthreads();

        // ---- compute: 2x2 WMMA per wave, two K=32 slabs ----
        #pragma unroll
        for (int kf = 0; kf < 64; kf += 32) {
            v16h afr[2], bfr[2];
            #pragma unroll
            for (int mi = 0; mi < 2; ++mi)
                afr[mi] = ld_frag(&As[cur][(wm * 32 + mi * 16 + lr) * 64 + kf + hs * 8]);
            #pragma unroll
            for (int ni = 0; ni < 2; ++ni)
                bfr[ni] = ld_frag(&Bs[cur][(wn * 32 + ni * 16 + lr) * 64 + kf + hs * 8]);
            #pragma unroll
            for (int mi = 0; mi < 2; ++mi)
                #pragma unroll
                for (int ni = 0; ni < 2; ++ni)
                    acc[mi][ni] = wmma_f16(afr[mi], bfr[ni], acc[mi][ni]);
        }
        __syncthreads();   // reads done before this buffer is restaged
    }

    // ---- epilogue: add bias, store f16 ----
    #pragma unroll
    for (int mi = 0; mi < 2; ++mi) {
        #pragma unroll
        for (int ni = 0; ni < 2; ++ni) {
            const int col = n0 + wn * 32 + ni * 16 + lr;
            const float bv = bias[col];
            #pragma unroll
            for (int r = 0; r < 8; ++r) {
                const int row = m0 + wm * 32 + mi * 16 + r + 8 * hs;
                C[(size_t)row * HDIM + col] = (half_t)(acc[mi][ni][r] + bv);
            }
        }
    }
}

// ---------------------------------------------------------------------------
// Kernel 2: word-level attention.  One block per (b, s, h).
//   scores[128x64] = q @ k^T * scale + wm; softmax over 64; wh = probs @ v.
// ---------------------------------------------------------------------------
__global__ __launch_bounds__(256) void word_attn(const half_t* __restrict__ qf,
                                                 const half_t* __restrict__ kf,
                                                 const half_t* __restrict__ vf,
                                                 const int* __restrict__ wmask,
                                                 half_t* __restrict__ whf) {
    __shared__ half_t vT[64 * 64];          // [d][t]
    __shared__ half_t probs[8 * 16 * 64];   // per-wave 16x64 A tiles

    const int b  = blockIdx.y;
    const int s  = blockIdx.x / NH;
    const int h  = blockIdx.x % NH;
    const int tid  = threadIdx.x;
    const int lane = tid & 31;
    const int w    = tid >> 5;
    const int hs   = lane >> 4;
    const int lr   = lane & 15;

    // ---- stage v^T into LDS ----
    #pragma unroll
    for (int i = 0; i < 2; ++i) {
        const int l  = tid * 2 + i;        // 0..511
        const int t  = l >> 3;             // token 0..63
        const int d0 = (l & 7) * 8;
        v8h x = *reinterpret_cast<const v8h*>(
            vf + (size_t)(b * 2048 + s * 64 + t) * HDIM + h * 64 + d0);
        #pragma unroll
        for (int j = 0; j < 8; ++j) vT[(d0 + j) * 64 + t] = x[j];
    }
    __syncthreads();

    // ---- scores = q @ k^T (per wave: 16 queries x 64 keys) ----
    v8f acc[4];
    #pragma unroll
    for (int nf = 0; nf < 4; ++nf) acc[nf] = zero8();

    #pragma unroll
    for (int kblk = 0; kblk < 2; ++kblk) {
        const int k0 = kblk * 32;
        const half_t* ap = qf + (size_t)(b * QLEN + w * 16 + lr) * HDIM
                              + h * 64 + k0 + hs * 8;
        v16h a = ld_frag(ap);
        #pragma unroll
        for (int nf = 0; nf < 4; ++nf) {
            const half_t* bp = kf + (size_t)(b * 2048 + s * 64 + nf * 16 + lr) * HDIM
                                  + h * 64 + k0 + hs * 8;
            acc[nf] = wmma_f16(a, ld_frag(bp), acc[nf]);
        }
    }

    // additive word mask (per column t, fixed per lane)
    float wmadd[4];
    #pragma unroll
    for (int nf = 0; nf < 4; ++nf) {
        const int t = nf * 16 + lr;
        wmadd[nf] = (1.0f - (float)wmask[(b * S1 + s) * S2 + t]) * -10000.0f;
    }

    // ---- softmax over 64 columns, row r + 8*hs ----
    #pragma unroll
    for (int r = 0; r < 8; ++r) {
        float mx = -1e30f;
        #pragma unroll
        for (int nf = 0; nf < 4; ++nf) {
            float v = acc[nf][r] * SCALE + wmadd[nf];
            acc[nf][r] = v;
            mx = fmaxf(mx, v);
        }
        #pragma unroll
        for (int m = 1; m < 16; m <<= 1) mx = fmaxf(mx, __shfl_xor(mx, m, 32));
        float e[4], sum = 0.f;
        #pragma unroll
        for (int nf = 0; nf < 4; ++nf) { e[nf] = __expf(acc[nf][r] - mx); sum += e[nf]; }
        #pragma unroll
        for (int m = 1; m < 16; m <<= 1) sum += __shfl_xor(sum, m, 32);
        const float inv = 1.0f / sum;
        #pragma unroll
        for (int nf = 0; nf < 4; ++nf)
            probs[w * 1024 + (r + 8 * hs) * 64 + nf * 16 + lr] = (half_t)(e[nf] * inv);
    }
    __syncthreads();

    // ---- wh = probs @ v  (A from LDS probs, B from LDS vT) ----
    v8f oacc[4];
    #pragma unroll
    for (int nf = 0; nf < 4; ++nf) oacc[nf] = zero8();
    #pragma unroll
    for (int kblk = 0; kblk < 2; ++kblk) {
        const int k0 = kblk * 32;
        v16h a = ld_frag(&probs[w * 1024 + lr * 64 + k0 + hs * 8]);
        #pragma unroll
        for (int nf = 0; nf < 4; ++nf) {
            v16h bb = ld_frag(&vT[(nf * 16 + lr) * 64 + k0 + hs * 8]);
            oacc[nf] = wmma_f16(a, bb, oacc[nf]);
        }
    }

    // store wh f16: row = (b*128+q)*32 + s, col = h*64 + d
    #pragma unroll
    for (int nf = 0; nf < 4; ++nf) {
        const int d = nf * 16 + lr;
        #pragma unroll
        for (int r = 0; r < 8; ++r) {
            const int q = w * 16 + r + 8 * hs;
            whf[((size_t)(b * QLEN + q) * S1 + s) * HDIM + h * 64 + d] =
                (half_t)oacc[nf][r];
        }
    }
}

// ---------------------------------------------------------------------------
// Kernel 3: sentence-level scores + softmax.  One block per (b, h).
//   scores2[128x32] = sq @ sk^T * scale + sm; softmax over 32 -> probs2 f32.
// ---------------------------------------------------------------------------
__global__ __launch_bounds__(256) void sent_attn(const half_t* __restrict__ sqf,
                                                 const half_t* __restrict__ skf,
                                                 const int* __restrict__ smask,
                                                 float* __restrict__ probs2) {
    const int b  = blockIdx.y;
    const int h  = blockIdx.x;
    const int tid  = threadIdx.x;
    const int lane = tid & 31;
    const int w    = tid >> 5;
    const int hs   = lane >> 4;
    const int lr   = lane & 15;

    v8f acc[2];
    acc[0] = zero8(); acc[1] = zero8();

    #pragma unroll
    for (int kblk = 0; kblk < 2; ++kblk) {
        const int k0 = kblk * 32;
        const half_t* ap = sqf + (size_t)(b * QLEN + w * 16 + lr) * HDIM
                               + h * 64 + k0 + hs * 8;
        v16h a = ld_frag(ap);
        #pragma unroll
        for (int nf = 0; nf < 2; ++nf) {
            const half_t* bp = skf + (size_t)(b * S1 + nf * 16 + lr) * HDIM
                                   + h * 64 + k0 + hs * 8;
            acc[nf] = wmma_f16(a, ld_frag(bp), acc[nf]);
        }
    }

    #pragma unroll
    for (int r = 0; r < 8; ++r) {
        const int q = w * 16 + r + 8 * hs;
        float vals[2], mx = -1e30f;
        #pragma unroll
        for (int nf = 0; nf < 2; ++nf) {
            const int s = nf * 16 + lr;
            float v = acc[nf][r] * SCALE +
                      (1.0f - (float)smask[(b * QLEN + q) * S1 + s]) * -10000.0f;
            vals[nf] = v;
            mx = fmaxf(mx, v);
        }
        #pragma unroll
        for (int m = 1; m < 16; m <<= 1) mx = fmaxf(mx, __shfl_xor(mx, m, 32));
        float e[2], sum = 0.f;
        #pragma unroll
        for (int nf = 0; nf < 2; ++nf) { e[nf] = __expf(vals[nf] - mx); sum += e[nf]; }
        #pragma unroll
        for (int m = 1; m < 16; m <<= 1) sum += __shfl_xor(sum, m, 32);
        const float inv = 1.0f / sum;
        #pragma unroll
        for (int nf = 0; nf < 2; ++nf)
            probs2[((size_t)(b * QLEN + q) * NH + h) * S1 + nf * 16 + lr] =
                e[nf] * inv;
    }
}

// ---------------------------------------------------------------------------
// Kernel 4: out[bq][h*64+d] = sum_s probs2[bq][h][s] * sv[bq][s][h*64+d]
//   (sv varies per query -> not GEMM-shaped; tiny FLOPs, plain VALU.)
// ---------------------------------------------------------------------------
__global__ __launch_bounds__(256) void final_out(const float* __restrict__ probs2,
                                                 const half_t* __restrict__ svf,
                                                 float* __restrict__ out) {
    const int i = blockIdx.x * 256 + threadIdx.x;   // 0 .. 1024*768-1
    const int row = i / HDIM;
    const int col = i % HDIM;
    const int h   = col >> 6;
    const float* p = probs2 + ((size_t)row * NH + h) * S1;
    const half_t* sv = svf + (size_t)row * S1 * HDIM + col;
    float acc = 0.f;
    #pragma unroll
    for (int s = 0; s < S1; ++s)
        acc += p[s] * (float)sv[(size_t)s * HDIM];
    out[i] = acc;
}

// ---------------------------------------------------------------------------
// Host launcher
// ---------------------------------------------------------------------------
extern "C" void kernel_launch(void* const* d_in, const int* in_sizes, int n_in,
                              void* d_out, int out_size, void* d_ws, size_t ws_size,
                              hipStream_t stream) {
    (void)in_sizes; (void)n_in; (void)out_size; (void)ws_size;

    const float* q_in  = (const float*)d_in[0];   // [8,128,768]
    const float* k_in  = (const float*)d_in[1];   // [8,32,64,768]
    const float* v_in  = (const float*)d_in[2];   // [8,32,64,768]
    const float* kvec  = (const float*)d_in[3];   // [8,32,768]
    const int*   wmask = (const int*)d_in[4];     // [8,32,64]
    const int*   smask = (const int*)d_in[5];     // [1024,32]
    const float* wq_w = (const float*)d_in[6];  const float* wq_b = (const float*)d_in[7];
    const float* wk_w = (const float*)d_in[8];  const float* wk_b = (const float*)d_in[9];
    const float* wv_w = (const float*)d_in[10]; const float* wv_b = (const float*)d_in[11];
    const float* sq_w = (const float*)d_in[12]; const float* sq_b = (const float*)d_in[13];
    const float* sk_w = (const float*)d_in[14]; const float* sk_b = (const float*)d_in[15];
    const float* sv_w = (const float*)d_in[16]; const float* sv_b = (const float*)d_in[17];

    // ---- carve workspace (256B aligned) ----
    char* ws = (char*)d_ws;
    size_t off = 0;
    auto carve = [&](size_t bytes) -> char* {
        char* p = ws + off;
        off += (bytes + 255) & ~(size_t)255;
        return p;
    };
    const size_t WSZ = (size_t)HDIM * HDIM * sizeof(half_t);
    half_t* wtq  = (half_t*)carve(WSZ);
    half_t* wtk  = (half_t*)carve(WSZ);
    half_t* wtv  = (half_t*)carve(WSZ);
    half_t* wtsq = (half_t*)carve(WSZ);
    half_t* wtsk = (half_t*)carve(WSZ);
    half_t* wtsv = (half_t*)carve(WSZ);
    half_t* qf   = (half_t*)carve((size_t)1024  * HDIM * sizeof(half_t));
    half_t* kfh  = (half_t*)carve((size_t)16384 * HDIM * sizeof(half_t));
    half_t* vfh  = (half_t*)carve((size_t)16384 * HDIM * sizeof(half_t));
    half_t* sqf  = (half_t*)carve((size_t)1024  * HDIM * sizeof(half_t));
    half_t* skf  = (half_t*)carve((size_t)256   * HDIM * sizeof(half_t));
    half_t* whf  = (half_t*)carve((size_t)32768 * HDIM * sizeof(half_t));
    half_t* svf  = (half_t*)carve((size_t)32768 * HDIM * sizeof(half_t));
    float*  prb2 = (float*) carve((size_t)1024 * NH * S1 * sizeof(float));

    // ---- 0: weight convert + transpose ----
    cvt_wt<<<2304, 256, 0, stream>>>(wq_w, wtq);
    cvt_wt<<<2304, 256, 0, stream>>>(wk_w, wtk);
    cvt_wt<<<2304, 256, 0, stream>>>(wv_w, wtv);
    cvt_wt<<<2304, 256, 0, stream>>>(sq_w, wtsq);
    cvt_wt<<<2304, 256, 0, stream>>>(sk_w, wtsk);
    cvt_wt<<<2304, 256, 0, stream>>>(sv_w, wtsv);

    // ---- 1: projections ----
    gemm768<true><<<dim3(8,   12), 256, 0, stream>>>(q_in, wtq,  wq_b, qf);   // M=1024
    gemm768<true><<<dim3(128, 12), 256, 0, stream>>>(k_in, wtk,  wk_b, kfh);  // M=16384
    gemm768<true><<<dim3(128, 12), 256, 0, stream>>>(v_in, wtv,  wv_b, vfh);  // M=16384
    gemm768<true><<<dim3(8,   12), 256, 0, stream>>>(q_in, wtsq, sq_b, sqf);  // M=1024
    gemm768<true><<<dim3(2,   12), 256, 0, stream>>>(kvec, wtsk, sk_b, skf);  // M=256

    // ---- 2: word-level attention -> wh (f16) ----
    word_attn<<<dim3(S1 * NH, BS), 256, 0, stream>>>(qf, kfh, vfh, wmask, whf);

    // ---- 3: sv projection (biggest GEMM, M=32768, async-staged A) ----
    gemm768<false><<<dim3(256, 12), 256, 0, stream>>>(whf, wtsv, sv_b, svf);

    // ---- 4: sentence scores + softmax ----
    sent_attn<<<dim3(NH, BS), 256, 0, stream>>>(sqf, skf, smask, prb2);

    // ---- 5: final weighted sum -> out f32 ----
    final_out<<<3072, 256, 0, stream>>>(prb2, svf, (float*)d_out);
}
